// BasisNetwork_18305150616436
// MI455X (gfx1250) — compile-verified, matrix operand order checked by reading, not run on hardware
//
#include <hip/hip_runtime.h>
#include <stdint.h>

// BasisNetwork (continuous conv GNN) for MI455X / gfx1250.
//
// Restructured: msg[e] = sum_b phi[e,b] * (X @ W_b)[edge_j[e]].
//  - Node-side GEMM Y[N,544] = X[N,32] @ [W_0|...|W_15|W_fc][32,544]
//    done with V_WMMA_F32_16X16X32_F16 (K=32 == feature width, 1 WMMA/tile).
//  - Hat basis => exactly 4 of 16 phi entries nonzero per edge; edge kernel
//    (one wave32 per edge, lane = channel) gathers 4x128B from Y (L2-resident,
//    43.5MB << 192MB L2) and atomicAdd-scatters 128B per edge.
//  - FC branch folded into the same GEMM as columns 512..543.

typedef __attribute__((ext_vector_type(16))) _Float16 v16h;
typedef __attribute__((ext_vector_type(8)))  float    v8f;

#define NLAYERS 4
#define FDIM    32
#define BSZ     16          // 4x4 tensor-product hat basis
#define NCOLS   544         // 16*32 conv cols + 32 fc cols
#define NTILES  34          // NCOLS / 16

struct PtrQuad { const float* p[NLAYERS]; };

static inline size_t align256(size_t x) { return (x + 255) & ~(size_t)255; }

// ---------------- phi compaction: per edge store (ix,iy,mask) + (fx,fy) -----
// hat_k(d) on centers linspace(-1,1,4): with t=(clamp(d)+1)*1.5 in [0,3],
// hat_{i0}=1-frac, hat_{i0+1}=frac, i0=min(floor(t),2). Partition of unity.
__global__ void k_phi(const float* __restrict__ attr, const int* __restrict__ ei,
                      const int* __restrict__ ej, uint32_t* __restrict__ meta,
                      float2* __restrict__ frac, int E) {
    int e = blockIdx.x * blockDim.x + threadIdx.x;
    if (e >= E) return;
    float dx = fminf(fmaxf(attr[2 * e + 0], -1.f), 1.f);
    float dy = fminf(fmaxf(attr[2 * e + 1], -1.f), 1.f);
    float tx = (dx + 1.f) * 1.5f;
    float ty = (dy + 1.f) * 1.5f;
    float ixf = fminf(floorf(tx), 2.f);
    float iyf = fminf(floorf(ty), 2.f);
    float2 f; f.x = tx - ixf; f.y = ty - iyf;
    frac[e] = f;
    uint32_t mask = (ei[e] != ej[e]) ? 0x10000u : 0u;   // centerIgnore
    meta[e] = (uint32_t)ixf | ((uint32_t)iyf << 8) | mask;
}

// ---------------- weight pack: fp32 -> f16, WMMA B-matrix VGPR layout -------
// wp[layer][ntile][lane][h]: lane = column (nt*16 + lane%16),
// halves h: K = (lane/16)*8 + (h<8 ? h : h+8)   (16-bit operand layout).
__global__ void k_pack_w(PtrQuad cw, PtrQuad fw, _Float16* __restrict__ wp) {
    int t = blockIdx.x * blockDim.x + threadIdx.x;
    if (t >= NLAYERS * NTILES * 32) return;
    int lane = t & 31;
    int nt   = (t >> 5) % NTILES;
    int l    = t / (NTILES * 32);
    const float* cwl = cw.p[l];
    const float* fwl = fw.p[l];
    int col = nt * 16 + (lane & 15);
    int kb  = (lane >> 4) << 3;
    _Float16* dst = wp + (size_t)t * 16;
#pragma unroll
    for (int h = 0; h < 16; ++h) {
        int k = kb + (h < 8 ? h : h + 8);
        float v;
        if (col < BSZ * FDIM) {
            int b = col >> 5, co = col & 31;                 // W[b][k][co]
            v = cwl[(b * FDIM + k) * FDIM + co];
        } else {
            v = fwl[k * FDIM + (col - BSZ * FDIM)];          // fc_w[k][co]
        }
        dst[h] = (_Float16)v;
    }
}

// ---------------- fp32 -> f16 activation cast (optional ReLU) ---------------
__global__ void k_cast(const float* __restrict__ x, _Float16* __restrict__ a,
                       int cnt, int relu) {
    int t = blockIdx.x * blockDim.x + threadIdx.x;
    if (t >= cnt) return;
    float v = x[t];
    if (relu) v = fmaxf(v, 0.f);
    a[t] = (_Float16)v;
}

// ---------------- WMMA GEMM: Y[N,544] = act[N,32] @ W[32,544] ---------------
// One wave per 16x16 output tile; K=32 -> exactly one v_wmma_f32_16x16x32_f16.
__global__ void k_gemm(const _Float16* __restrict__ act,
                       const _Float16* __restrict__ wp,
                       float* __restrict__ Y, int mtiles) {
    int wid  = (blockIdx.x * blockDim.x + threadIdx.x) >> 5;
    int lane = threadIdx.x & 31;
    int mt = wid / NTILES;
    int nt = wid - mt * NTILES;
    if (mt >= mtiles) return;

    // A operand (16x32 f16): lane<16 -> row M=lane, K {0..7,16..23};
    //                        lane>=16 -> row M=lane-16, K {8..15,24..31}
    int row = (mt << 4) + (lane & 15);
    int kb  = (lane >> 4) << 3;
    const _Float16* ap = act + (size_t)row * FDIM;
    v16h a;
    ((float4*)&a)[0] = *(const float4*)(ap + kb);        // K kb..kb+7
    ((float4*)&a)[1] = *(const float4*)(ap + kb + 16);   // K kb+16..kb+23

    // B operand: pre-packed, one contiguous 32B load per lane
    v16h b = *(const v16h*)(wp + ((size_t)(nt * 32 + lane)) * 16);

    v8f c = {};
    c = __builtin_amdgcn_wmma_f32_16x16x32_f16(
            /*neg_a=*/false, a, /*neg_b=*/false, b,
            /*c_mod=*/(short)0, c, /*reuse_a=*/false, /*reuse_b=*/false);

    // D layout: VGPR v -> M = v + (lane<16 ? 0 : 8), N = lane%16
    int n  = (nt << 4) + (lane & 15);
    int m0 = (mt << 4) + ((lane >> 4) << 3);
    float* yp = Y + (size_t)m0 * NCOLS + n;
#pragma unroll
    for (int v = 0; v < 8; ++v) yp[(size_t)v * NCOLS] = c[v];
}

// ---------------- node init: ans (+)= Y_fc + bias ---------------------------
__global__ void k_node(const float* __restrict__ Y, const float* __restrict__ bias,
                       float* __restrict__ ans, int cnt, int first) {
    int t = blockIdx.x * blockDim.x + threadIdx.x;
    if (t >= cnt) return;
    int n = t >> 5, f = t & 31;
    float v = Y[(size_t)n * NCOLS + BSZ * FDIM + f] + bias[f];
    ans[t] = first ? v : (ans[t] + v);
}

// ---------------- edge conv: gather 4 basis blocks from Y, scatter-add ------
// One wave32 per edge; lane = output channel.
__global__ void k_edge(const int* __restrict__ ei, const int* __restrict__ ej,
                       const uint32_t* __restrict__ meta,
                       const float2* __restrict__ frac,
                       const float* __restrict__ Y,
                       float* __restrict__ ans, int E) {
    int e    = (blockIdx.x * blockDim.x + threadIdx.x) >> 5;
    int lane = threadIdx.x & 31;
    if (e >= E) return;
    uint32_t m = meta[e];
    if (!(m & 0x10000u)) return;                 // self-edge: zero contribution
    int ix = m & 3, iy = (m >> 8) & 3;
    float2 f = frac[e];
    float fx = f.x, fy = f.y;
    float w00 = (1.f - fx) * (1.f - fy);
    float w01 = (1.f - fx) * fy;
    float w10 = fx * (1.f - fy);
    float w11 = fx * fy;
    int b00 = ix * 4 + iy;                        // phi index = bx*4 + by
    const float* yj = Y + (size_t)ej[e] * NCOLS;
    float msg = w00 * yj[(b00    ) * FDIM + lane]
              + w01 * yj[(b00 + 1) * FDIM + lane]
              + w10 * yj[(b00 + 4) * FDIM + lane]
              + w11 * yj[(b00 + 5) * FDIM + lane];
    atomicAdd(&ans[(size_t)ei[e] * FDIM + lane], msg);
}

// ---------------- finalize: out = ans / 128 ---------------------------------
__global__ void k_final(const float* __restrict__ ans, float* __restrict__ out,
                        int cnt) {
    int t = blockIdx.x * blockDim.x + threadIdx.x;
    if (t >= cnt) return;
    out[t] = ans[t] * (1.0f / 128.0f);
}

extern "C" void kernel_launch(void* const* d_in, const int* in_sizes, int n_in,
                              void* d_out, int out_size, void* d_ws, size_t ws_size,
                              hipStream_t stream) {
    const float* x    = (const float*)d_in[0];
    const int*   ei   = (const int*)d_in[1];
    const int*   ej   = (const int*)d_in[2];
    const float* attr = (const float*)d_in[3];
    const int N = in_sizes[0] / FDIM;
    const int E = in_sizes[1];

    PtrQuad cw, fw, fb;
    if (n_in >= 16) {
        // lists flattened into separate entries
        for (int l = 0; l < NLAYERS; ++l) {
            cw.p[l] = (const float*)d_in[4 + l];
            fw.p[l] = (const float*)d_in[8 + l];
            fb.p[l] = (const float*)d_in[12 + l];
        }
    } else {
        // lists concatenated per dict entry
        const float* c0 = (const float*)d_in[4];
        const float* f0 = (const float*)d_in[5];
        const float* b0 = (const float*)d_in[6];
        for (int l = 0; l < NLAYERS; ++l) {
            cw.p[l] = c0 + (size_t)l * BSZ * FDIM * FDIM;
            fw.p[l] = f0 + (size_t)l * FDIM * FDIM;
            fb.p[l] = b0 + (size_t)l * FDIM;
        }
    }

    // -------- workspace carve (~51 MB total) --------
    char* w = (char*)d_ws;
    _Float16* wpack = (_Float16*)w; w += align256((size_t)NLAYERS * NTILES * 32 * 16 * 2);
    uint32_t* meta  = (uint32_t*)w; w += align256((size_t)E * 4);
    float2*   frac  = (float2*)w;   w += align256((size_t)E * 8);
    _Float16* act   = (_Float16*)w; w += align256((size_t)N * FDIM * 2);
    float*    Y     = (float*)w;    w += align256((size_t)N * NCOLS * 4);
    float*    ans   = (float*)w;

    const int mtiles = (N + 15) / 16;
    const int nodecnt = N * FDIM;
    const int gemm_waves = mtiles * NTILES;
    const int gemm_blocks = (gemm_waves * 32 + 255) / 256;

    // one-time prep (layer-invariant)
    k_pack_w<<<(NLAYERS * NTILES * 32 + 255) / 256, 256, 0, stream>>>(cw, fw, wpack);
    k_phi<<<(E + 255) / 256, 256, 0, stream>>>(attr, ei, ej, meta, frac, E);
    k_cast<<<(nodecnt + 255) / 256, 256, 0, stream>>>(x, act, nodecnt, 0);

    for (int l = 0; l < NLAYERS; ++l) {
        const _Float16* wpl = wpack + (size_t)l * NTILES * 32 * 16;
        k_gemm<<<gemm_blocks, 256, 0, stream>>>(act, wpl, Y, mtiles);
        k_node<<<(nodecnt + 255) / 256, 256, 0, stream>>>(Y, fb.p[l], ans, nodecnt, l == 0);
        k_edge<<<(E + 7) / 8, 256, 0, stream>>>(ei, ej, meta, frac, Y, ans, E);
        if (l + 1 < NLAYERS)
            k_cast<<<(nodecnt + 255) / 256, 256, 0, stream>>>(ans, act, nodecnt, 1);
    }

    k_final<<<(out_size + 255) / 256, 256, 0, stream>>>(ans, (float*)d_out, out_size);
}
